// GatedDeltaNetVarlen_59064390255183
// MI455X (gfx1250) — compile-verified, matrix operand order checked by reading, not run on hardware
//
#include <hip/hip_runtime.h>
#include <hip/hip_bf16.h>

// ---------------------------------------------------------------------------
// GatedDeltaNet varlen forward for MI455X (gfx1250, wave32, WMMA).
//   1) gemm<8192,2048>: mixed = hs @ w_qkv
//   2) gemm<4096,2048>: z     = hs @ w_z
//   3) ba_kernel:       beta = sigmoid(hs@w_b), g = -exp(A_log)*softplus(hs@w_a+dt)
//   4) conv_kernel:     causal depthwise conv(K=4) + SiLU + split + l2norm(q,k)
//   5) delta_kernel:    gated delta rule scan, column-parallel state in VGPRs
//   6) normgate:        RMSNorm(o)*norm_w * SiLU(z)
//   7) gemm<2048,4096>: out = gated @ w_out
// GEMMs: bf16 WMMA, 128x128 block tile, K-tile 32, double-buffered LDS with
// register staging so global loads of tile kt+1 overlap WMMA on tile kt.
// ---------------------------------------------------------------------------

#define HID 2048
#define NK 16
#define NV 32
#define DK 128
#define DV 128
#define KSZ 4
#define KEY_DIM (NK*DK)      // 2048
#define VAL_DIM (NV*DV)      // 4096
#define CONV_DIM (2*KEY_DIM + VAL_DIM)  // 8192
#define TOTAL_T 4096
#define NSEQ 4

typedef __attribute__((ext_vector_type(16))) __bf16 v16bf;
typedef __attribute__((ext_vector_type(8)))  float  v8f;

// pack 4 f32 -> 4 bf16 and store as one 8-byte LDS write (ds_store_b64)
__device__ __forceinline__ void st4bf(__bf16* dst, float x, float y, float z, float w) {
  union { __bf16 h[4]; unsigned long long u; } u;
  u.h[0] = (__bf16)x; u.h[1] = (__bf16)y; u.h[2] = (__bf16)z; u.h[3] = (__bf16)w;
  *(unsigned long long*)dst = u.u;
}

// ---------------------------------------------------------------------------
// C[M,N] = A[M,K] @ B[K,N]  (f32 in memory, bf16 WMMA math).
// N,K compile-time so all strides/offsets fold to immediates.
// 256 threads = 8 waves (4Mx2N); each wave owns 2x4 16x16 accumulators.
// All dims divide tiles exactly -> EXEC all-ones (WMMA requirement).
// ---------------------------------------------------------------------------
template<int N, int K>
__global__ __launch_bounds__(256) void gemm_bf16_wmma(
    const float* __restrict__ A, const float* __restrict__ B,
    float* __restrict__ C)
{
  __shared__ __align__(16) __bf16 Al[2][128][40];  // A tile [m][k], bf16
  __shared__ __align__(16) __bf16 Bt[2][128][40];  // B tile transposed [n][k]

  const int tid  = threadIdx.x;
  const int lane = tid & 31;
  const int wave = tid >> 5;
  const int m0 = blockIdx.y * 128;
  const int n0 = blockIdx.x * 128;
  const int wm = (wave >> 1) * 32;
  const int wn = (wave & 1) * 64;

  // --- per-thread tile-load geometry (it = 0..3 slots) ---
  const int arow = tid >> 3;          // + it*32  (A row in tile)
  const int acol = (tid & 7) * 4;     // A k-offset in tile (float4)
  const int bn   = tid & 127;         // B column n in tile (fixed)
  const int bk   = (tid >> 7) * 4;    // + it*8   (B k-offset in tile)

  const float* Ap = A + (size_t)(m0 + arow) * K + acol;  // advance by 32/iter
  const float* Bp = B + (size_t)bk * N + n0 + bn;        // advance by 32*N/iter

  v8f acc[2][4] = {};
  const int r15 = lane & 15;
  const int kb  = (lane < 16) ? 0 : 8;   // K-half per CDNA5 16-bit WMMA layout

  float4 ra[4];        // A staging: 4 rows x float4
  float  rb[4][4];     // B staging: 4 k-slots x 4 consecutive k

  // prologue: stage tile 0
  #pragma unroll
  for (int it = 0; it < 4; ++it)
    ra[it] = *(const float4*)(Ap + (size_t)it * 32 * K);
  #pragma unroll
  for (int it = 0; it < 4; ++it)
    #pragma unroll
    for (int u = 0; u < 4; ++u)
      rb[it][u] = Bp[(size_t)(it * 8 + u) * N];

  int buf = 0;
  for (int kt = 0; kt < K; kt += 32) {
    // commit staged tile to LDS (packed b64 stores)
    #pragma unroll
    for (int it = 0; it < 4; ++it)
      st4bf(&Al[buf][arow + it*32][acol], ra[it].x, ra[it].y, ra[it].z, ra[it].w);
    #pragma unroll
    for (int it = 0; it < 4; ++it)
      st4bf(&Bt[buf][bn][bk + it*8], rb[it][0], rb[it][1], rb[it][2], rb[it][3]);
    __syncthreads();

    // stage next tile while WMMAs run (loads in flight during compute)
    if (kt + 32 < K) {
      const float* Ap2 = Ap + kt + 32;
      const float* Bp2 = Bp + (size_t)(kt + 32) * N;
      #pragma unroll
      for (int it = 0; it < 4; ++it)
        ra[it] = *(const float4*)(Ap2 + (size_t)it * 32 * K);
      #pragma unroll
      for (int it = 0; it < 4; ++it)
        #pragma unroll
        for (int u = 0; u < 4; ++u)
          rb[it][u] = Bp2[(size_t)(it * 8 + u) * N];
    }

    // fragments per CDNA5 16-bit layout: lanes 0-15 K 0-7/16-23, 16-31 K 8-15/24-31
    v16bf af[2], bfr[4];
    #pragma unroll
    for (int i = 0; i < 2; ++i) {
      const __bf16* p = &Al[buf][wm + i*16 + r15][kb];
      #pragma unroll
      for (int e = 0; e < 8; ++e) { af[i][e] = p[e]; af[i][8+e] = p[16+e]; }
    }
    #pragma unroll
    for (int j = 0; j < 4; ++j) {
      const __bf16* p = &Bt[buf][wn + j*16 + r15][kb];
      #pragma unroll
      for (int e = 0; e < 8; ++e) { bfr[j][e] = p[e]; bfr[j][8+e] = p[16+e]; }
    }
    #pragma unroll
    for (int i = 0; i < 2; ++i)
      #pragma unroll
      for (int j = 0; j < 4; ++j)
        acc[i][j] = __builtin_amdgcn_wmma_f32_16x16x32_bf16(
            false, af[i], false, bfr[j], (short)0, acc[i][j], false, false);
    buf ^= 1;
  }

  // epilogue: one 64-bit base, all offsets compile-time immediates
  const int rowbase = m0 + wm + ((lane < 16) ? 0 : 8);
  const int colbase = n0 + wn + r15;
  float* Cp = C + (size_t)rowbase * N + colbase;
  #pragma unroll
  for (int i = 0; i < 2; ++i)
    #pragma unroll
    for (int j = 0; j < 4; ++j)
      #pragma unroll
      for (int r = 0; r < 8; ++r)
        Cp[(i*16 + r) * N + j*16] = acc[i][j][r];
}

// ---------------------------------------------------------------------------
// beta/g projections: tiny GEMV (64 outputs) fused with activations.
// ---------------------------------------------------------------------------
__global__ __launch_bounds__(256) void ba_kernel(
    const float* __restrict__ hs, const float* __restrict__ w_b,
    const float* __restrict__ w_a, const float* __restrict__ dt_bias,
    const float* __restrict__ A_log,
    float* __restrict__ beta, float* __restrict__ g)
{
  const int t = blockIdx.x, tid = threadIdx.x;
  __shared__ float h[HID];
  __shared__ float part[256];
  for (int i = tid; i < HID; i += 256) h[i] = hs[(size_t)t * HID + i];
  __syncthreads();
  const int col = tid & 63;
  const int seg = tid >> 6;
  const float* w = (col < 32) ? (w_b + col) : (w_a + (col - 32));
  float acc = 0.f;
  for (int i = seg * 512; i < seg * 512 + 512; ++i) acc += h[i] * w[(size_t)i * 32];
  part[tid] = acc;
  __syncthreads();
  if (tid < 64) {
    float v = part[tid] + part[tid + 64] + part[tid + 128] + part[tid + 192];
    if (col < 32) {
      beta[t * NV + col] = 1.f / (1.f + __expf(-v));
    } else {
      int j = col - 32;
      float x = v + dt_bias[j];
      float sp = (x > 20.f) ? x : log1pf(__expf(x));
      g[t * NV + j] = -__expf(A_log[j]) * sp;
    }
  }
}

// ---------------------------------------------------------------------------
// Causal depthwise conv (K=4, seq-boundary masked) + SiLU + split + l2norm.
// ---------------------------------------------------------------------------
__global__ __launch_bounds__(256) void conv_kernel(
    const float* __restrict__ mixed, const float* __restrict__ conv_w,
    const int* __restrict__ cu,
    float* __restrict__ qn, float* __restrict__ kn, float* __restrict__ vb)
{
  const int t = blockIdx.x, tid = threadIdx.x;
  __shared__ float part[256];
  __shared__ float invn[32];

  int s = 0;
  #pragma unroll
  for (int i = 1; i <= NSEQ; ++i) s += (t >= cu[i]) ? 1 : 0;
  const int start = cu[s];
  const float m1 = (t - 1 >= start) ? 1.f : 0.f;
  const float m2 = (t - 2 >= start) ? 1.f : 0.f;
  const float m3 = (t - 3 >= start) ? 1.f : 0.f;
  const int tm1 = (t - 1 >= 0) ? t - 1 : 0;
  const int tm2 = (t - 2 >= 0) ? t - 2 : 0;
  const int tm3 = (t - 3 >= 0) ? t - 3 : 0;

  const int cbase = tid * 32;
  float res[32];
  float ss = 0.f;
  #pragma unroll
  for (int i = 0; i < 32; ++i) {
    const int c = cbase + i;
    const float x0 = mixed[(size_t)t   * CONV_DIM + c];
    const float x1 = mixed[(size_t)tm1 * CONV_DIM + c] * m1;
    const float x2 = mixed[(size_t)tm2 * CONV_DIM + c] * m2;
    const float x3 = mixed[(size_t)tm3 * CONV_DIM + c] * m3;
    const float4 wv = *(const float4*)(conv_w + (size_t)c * KSZ);
    float y = wv.w * x0 + wv.z * x1 + wv.y * x2 + wv.x * x3;
    y = y * (1.f / (1.f + __expf(-y)));   // SiLU
    res[i] = y;
    ss += y * y;
  }
  part[tid] = ss;
  __syncthreads();
  if (tid < 32) {
    float sum = part[tid*4] + part[tid*4+1] + part[tid*4+2] + part[tid*4+3];
    invn[tid] = rsqrtf(sum + 1e-6f);
  }
  __syncthreads();

  if (cbase < KEY_DIM) {
    const float iv = invn[cbase >> 7];
    #pragma unroll
    for (int i = 0; i < 32; ++i) qn[(size_t)t * KEY_DIM + cbase + i] = res[i] * iv;
  } else if (cbase < 2 * KEY_DIM) {
    const float iv = invn[cbase >> 7];
    #pragma unroll
    for (int i = 0; i < 32; ++i) kn[(size_t)t * KEY_DIM + cbase - KEY_DIM + i] = res[i] * iv;
  } else {
    #pragma unroll
    for (int i = 0; i < 32; ++i) vb[(size_t)t * VAL_DIM + cbase - 2*KEY_DIM + i] = res[i];
  }
}

// ---------------------------------------------------------------------------
// Gated delta rule scan: S[DK,DV] is column-separable; thread j owns column j
// in registers (zero cross-lane traffic). One block per (head, sequence).
// ---------------------------------------------------------------------------
__global__ __launch_bounds__(128) void delta_kernel(
    const float* __restrict__ qn, const float* __restrict__ kn,
    const float* __restrict__ vb, const float* __restrict__ beta,
    const float* __restrict__ g, const int* __restrict__ cu,
    float* __restrict__ o)
{
  const int h  = blockIdx.x;
  const int sq = blockIdx.y;
  const int j  = threadIdx.x;
  const int hk = h >> 1;        // GQA repeat factor 2
  const int t0 = cu[sq], t1 = cu[sq + 1];

  __shared__ float lk[DK];
  __shared__ float lq[DK];

  float S[DK];
  #pragma unroll
  for (int i = 0; i < DK; ++i) S[i] = 0.f;

  const float scale = 0.08838834764831845f;  // DK^-0.5

  for (int t = t0; t < t1; ++t) {
    __syncthreads();
    lk[j] = kn[((size_t)t * NK + hk) * DK + j];
    lq[j] = qn[((size_t)t * NK + hk) * DK + j];
    __syncthreads();

    const float eg = __expf(g[t * NV + h]);
    const float bt = beta[t * NV + h];
    const float vj = vb[((size_t)t * NV + h) * DV + j];

    float kv = 0.f;
    #pragma unroll
    for (int i = 0; i < DK; ++i) kv += lk[i] * S[i];
    kv *= eg;                               // k . (S * exp(g))
    const float w = bt * (vj - kv);

    float ot = 0.f;
    #pragma unroll
    for (int i = 0; i < DK; ++i) {
      S[i] = S[i] * eg + lk[i] * w;
      ot += lq[i] * S[i];
    }
    o[((size_t)t * NV + h) * DV + j] = ot * scale;
  }
}

// ---------------------------------------------------------------------------
// RMSNorm over DV + norm_w + SiLU(z) gate.
// ---------------------------------------------------------------------------
__global__ __launch_bounds__(256) void normgate_kernel(
    const float* __restrict__ o, const float* __restrict__ z,
    const float* __restrict__ norm_w, float* __restrict__ gated)
{
  const int t = blockIdx.x, tid = threadIdx.x;
  __shared__ float part[256];
  __shared__ float inv[32];
  const int cbase = tid * 16;
  float ov[16];
  float ss = 0.f;
  #pragma unroll
  for (int i = 0; i < 16; ++i) {
    float v = o[(size_t)t * VAL_DIM + cbase + i];
    ov[i] = v;
    ss += v * v;
  }
  part[tid] = ss;
  __syncthreads();
  if (tid < 32) {
    float s = 0.f;
    #pragma unroll
    for (int u = 0; u < 8; ++u) s += part[tid * 8 + u];
    inv[tid] = rsqrtf(s * (1.f / 128.f) + 1e-6f);
  }
  __syncthreads();
  const float iv = inv[cbase >> 7];
  #pragma unroll
  for (int i = 0; i < 16; ++i) {
    const int c = cbase + i;
    const float zz = z[(size_t)t * VAL_DIM + c];
    const float r = ov[i] * iv * norm_w[c & 127];
    gated[(size_t)t * VAL_DIM + c] = r * (zz * (1.f / (1.f + __expf(-zz))));
  }
}

// ---------------------------------------------------------------------------
extern "C" void kernel_launch(void* const* d_in, const int* in_sizes, int n_in,
                              void* d_out, int out_size, void* d_ws, size_t ws_size,
                              hipStream_t stream)
{
  const float* hs      = (const float*)d_in[0];
  const int*   cu      = (const int*)  d_in[1];
  const float* w_qkv   = (const float*)d_in[2];
  const float* w_z     = (const float*)d_in[3];
  const float* w_b     = (const float*)d_in[4];
  const float* w_a     = (const float*)d_in[5];
  const float* conv_w  = (const float*)d_in[6];
  const float* dt_bias = (const float*)d_in[7];
  const float* A_log   = (const float*)d_in[8];
  const float* norm_w  = (const float*)d_in[9];
  const float* w_out   = (const float*)d_in[10];
  float* out = (float*)d_out;

  char* ws = (char*)d_ws;
  size_t off = 0;
  auto carve = [&](size_t nfloats) { float* p = (float*)(ws + off); off += nfloats * sizeof(float); return p; };
  float* mixed = carve((size_t)TOTAL_T * CONV_DIM);   // 128 MB
  float* zbuf  = carve((size_t)TOTAL_T * VAL_DIM);    //  64 MB
  float* qn    = carve((size_t)TOTAL_T * KEY_DIM);    //  32 MB
  float* kn    = carve((size_t)TOTAL_T * KEY_DIM);    //  32 MB
  float* vb    = carve((size_t)TOTAL_T * VAL_DIM);    //  64 MB
  float* beta  = carve((size_t)TOTAL_T * NV);
  float* gb    = carve((size_t)TOTAL_T * NV);
  float* ob    = carve((size_t)TOTAL_T * VAL_DIM);    //  64 MB
  float* gated = mixed;  // mixed dead after conv_kernel; reuse

  const dim3 blk(256);

  gemm_bf16_wmma<CONV_DIM, HID><<<dim3(CONV_DIM/128, TOTAL_T/128), blk, 0, stream>>>(
      hs, w_qkv, mixed);
  gemm_bf16_wmma<VAL_DIM, HID><<<dim3(VAL_DIM/128, TOTAL_T/128), blk, 0, stream>>>(
      hs, w_z, zbuf);
  ba_kernel<<<TOTAL_T, blk, 0, stream>>>(hs, w_b, w_a, dt_bias, A_log, beta, gb);
  conv_kernel<<<TOTAL_T, blk, 0, stream>>>(mixed, conv_w, cu, qn, kn, vb);
  delta_kernel<<<dim3(NV, NSEQ), dim3(128), 0, stream>>>(qn, kn, vb, beta, gb, cu, ob);
  normgate_kernel<<<TOTAL_T, blk, 0, stream>>>(ob, zbuf, norm_w, gated);
  gemm_bf16_wmma<HID, VAL_DIM><<<dim3(HID/128, TOTAL_T/128), blk, 0, stream>>>(
      gated, w_out, out);
}
